// EncoderLayer_27023934227036
// MI455X (gfx1250) — matrix operand order repaired and checked by reading, not compiled, not run on hardware
//
#include <hip/hip_runtime.h>
#include <hip/hip_bf16.h>

typedef __attribute__((ext_vector_type(16))) _Float16 v16h;
typedef __attribute__((ext_vector_type(8)))  _Float16 v8h;
typedef __attribute__((ext_vector_type(8)))  float    v8f;

#define BATCH 8
#define MDIM  1024
#define SDIM  1024
#define NHEAD 8
#define DHEAD 128
static constexpr long PB = (long)MDIM * SDIM;          // 1048576 elems per batch slab
static constexpr long BHMM = (long)BATCH * NHEAD * PB; // attn elems

// ---------------------------------------------------------------------------
// Generic WMMA GEMM: C[z] = A[z] (Mt x K, row major) @ B[z] (K x Nt)
//   BT=true : B element (k,n) at Bp[n*ldb + k]   (weights stored [N,K], k_norm)
//   BT=false: B element (k,n) at Bp[k*ldb + n]   (V matrix)
// Wave tile 32x64: 2 A frags x 4 B frags -> 8 WMMAs per K-step; A frags
// reused 4x, B frags 2x from registers (1.5 b128 loads per WMMA).
// 8 waves/block in 4x2 arrangement -> 128x128 block tile.
// z decomposition: zb = z >> zshift, zh = z & ((1<<zshift)-1)  (batch / head)
// ---------------------------------------------------------------------------
struct GemmP {
  const _Float16* A;  long lda, aBatch, aHead;
  const _Float16* Bm; long ldb, bBatch, bHead;
  float* Cf; _Float16* Ch; long ldc, cBatch, cHead;
  const float* bias;
  const float* add;  long addBatch;
  const int* mask;   long maskBatch;
  int K; int zshift;
};

// EPI: 0 = store f16 | 2 = mask->-1e9, store f32 | 3 = +residual, store f32
//      4 = +bias, relu, store f16 | 5 = +bias +residual, store f32
template<bool BT, int EPI>
__global__ __launch_bounds__(256) void gemm_wmma_k(GemmP p) {
  const int lane = threadIdx.x & 31;
  const int wave = threadIdx.x >> 5;
  const int hi   = lane >> 4;          // lane 16..31 half
  const int lm   = lane & 15;
  const int wm   = wave >> 1;          // 4 wave rows
  const int wn   = wave & 1;           // 2 wave cols
  const long m0 = (long)blockIdx.y * 128 + wm * 32;
  const long n0 = (long)blockIdx.x * 128 + wn * 64;
  const long z  = blockIdx.z;
  const long zb = z >> p.zshift;
  const long zh = z - (zb << p.zshift);

  const _Float16* Ap = p.A  + zb * p.aBatch + zh * p.aHead + (m0 + lm) * p.lda;
  const _Float16* Bp = p.Bm + zb * p.bBatch + zh * p.bHead;

  v8f acc[2][4];
#pragma unroll
  for (int am = 0; am < 2; ++am)
#pragma unroll
    for (int bn = 0; bn < 4; ++bn) acc[am][bn] = (v8f){};

  for (int k0 = 0; k0 < p.K; k0 += 32) {
    // A 16x32 f16 fragments: lane=M; K = hi*16 + i (contiguous along K)
    v16h a[2];
#pragma unroll
    for (int am = 0; am < 2; ++am) {
      const _Float16* ap = Ap + (long)am * 16 * p.lda + k0 + hi * 16;
      v8h a0 = *(const v8h*)(ap);
      v8h a1 = *(const v8h*)(ap + 8);
#pragma unroll
      for (int i = 0; i < 8; ++i) { a[am][i] = a0[i]; a[am][i + 8] = a1[i]; }
    }
    // B 32x16 fragments: lane=N; K = hi*16 + i (contiguous along K)
    v16h b[4];
#pragma unroll
    for (int bn = 0; bn < 4; ++bn) {
      if (BT) {
        const _Float16* bp = Bp + (n0 + bn * 16 + lm) * p.ldb + k0 + hi * 16;
        v8h b0 = *(const v8h*)(bp);
        v8h b1 = *(const v8h*)(bp + 8);
#pragma unroll
        for (int i = 0; i < 8; ++i) { b[bn][i] = b0[i]; b[bn][i + 8] = b1[i]; }
      } else {
        const _Float16* bp = Bp + (long)(k0 + hi * 16) * p.ldb + (n0 + bn * 16 + lm);
#pragma unroll
        for (int i = 0; i < 16; ++i) b[bn][i] = bp[(long)i * p.ldb];
      }
    }
#pragma unroll
    for (int am = 0; am < 2; ++am)
#pragma unroll
      for (int bn = 0; bn < 4; ++bn)
        acc[am][bn] = __builtin_amdgcn_wmma_f32_16x16x32_f16(
            false, a[am], false, b[bn], (short)0, acc[am][bn], false, false);
  }

#pragma unroll
  for (int am = 0; am < 2; ++am) {
#pragma unroll
    for (int bn = 0; bn < 4; ++bn) {
      const long n = n0 + bn * 16 + lm;
#pragma unroll
      for (int r = 0; r < 8; ++r) {
        const long m = m0 + am * 16 + hi * 8 + r; // C/D: VGPR r -> M = r + 8*hi
        float v = acc[am][bn][r];
        const long cidx = zb * p.cBatch + zh * p.cHead + m * p.ldc + n;
        if (EPI == 2) {
          if (p.mask[zb * p.maskBatch + m * p.ldc + n] == 0) v = -1e9f;
          p.Cf[cidx] = v;
        } else if (EPI == 3) {
          p.Cf[cidx] = v + p.add[zb * p.addBatch + m * p.ldc + n];
        } else if (EPI == 4) {
          v += p.bias[n];
          p.Ch[cidx] = (_Float16)fmaxf(v, 0.0f);
        } else if (EPI == 5) {
          p.Cf[cidx] = v + p.bias[n] + p.add[zb * p.addBatch + m * p.ldc + n];
        } else {
          p.Ch[cidx] = (_Float16)v;
        }
      }
    }
  }
}

// ---------------------------------------------------------------------------
// LayerNorm over the full [1024,1024] slab per batch: stats pass
// ---------------------------------------------------------------------------
__global__ __launch_bounds__(256) void ln_stats_k(const float* __restrict__ x,
                                                  double* __restrict__ stats,
                                                  long n) {
  const float* xb = x + (long)blockIdx.x * n;
  double s = 0.0, ss = 0.0;
  for (long i = threadIdx.x; i < n; i += 256) {
    double v = (double)xb[i];
    s += v; ss += v * v;
  }
  __shared__ double sh[512];
  sh[threadIdx.x] = s; sh[256 + threadIdx.x] = ss;
  __syncthreads();
  for (int off = 128; off > 0; off >>= 1) {
    if ((int)threadIdx.x < off) {
      sh[threadIdx.x]       += sh[threadIdx.x + off];
      sh[256 + threadIdx.x] += sh[256 + threadIdx.x + off];
    }
    __syncthreads();
  }
  if (threadIdx.x == 0) {
    double mu  = sh[0] / (double)n;
    double var = sh[256] / (double)n - mu * mu;
    stats[2 * blockIdx.x]     = mu;
    stats[2 * blockIdx.x + 1] = 1.0 / sqrt(var + 1e-6);
  }
}

__global__ __launch_bounds__(256) void ln_apply_k(const float* __restrict__ x,
                                                  const double* __restrict__ stats,
                                                  const float* __restrict__ g,
                                                  const float* __restrict__ bb,
                                                  _Float16* __restrict__ out,
                                                  long perBatch, long total) {
  long stride = (long)gridDim.x * blockDim.x;
  for (long i = (long)blockIdx.x * blockDim.x + threadIdx.x; i < total; i += stride) {
    long b = i / perBatch;
    long j = i - b * perBatch;
    float mu = (float)stats[2 * b];
    float rs = (float)stats[2 * b + 1];
    out[i] = (_Float16)(((x[i] - mu) * rs) * g[j] + bb[j]);
  }
}

// F.normalize of K over the HEAD axis: for each (b,m,d), scale the 8 head vals
__global__ __launch_bounds__(256) void knorm_k(_Float16* __restrict__ k16) {
  long t = (long)blockIdx.x * blockDim.x + threadIdx.x;
  if (t >= (long)BATCH * MDIM * DHEAD) return;
  long b = t >> 17;                 // 1024*128 = 2^17
  long r = t & ((1L << 17) - 1);
  long m = r >> 7;
  long d = r & 127;
  _Float16* base = k16 + b * PB + m * SDIM + d;
  float vals[NHEAD];
  float s = 0.0f;
#pragma unroll
  for (int h = 0; h < NHEAD; ++h) { vals[h] = (float)base[h * DHEAD]; s += vals[h] * vals[h]; }
  float inv = 1.0f / fmaxf(sqrtf(s), 1e-12f);
#pragma unroll
  for (int h = 0; h < NHEAD; ++h) base[h * DHEAD] = (_Float16)(vals[h] * inv);
}

// Row softmax over 1024 cols; writes normalized f32 (final attn output) and
// an f16 copy used as the A operand of the attn @ V GEMM.
__global__ __launch_bounds__(256) void softmax_row_k(float* __restrict__ attn,
                                                     _Float16* __restrict__ attn16) {
  const long row = blockIdx.x;
  float* p = attn + row * 1024;
  __shared__ float sh[256];
  const int tid = threadIdx.x;
  float x[4];
  float mx = -3.4e38f;
#pragma unroll
  for (int j = 0; j < 4; ++j) { x[j] = p[tid + j * 256]; mx = fmaxf(mx, x[j]); }
  sh[tid] = mx; __syncthreads();
  for (int off = 128; off > 0; off >>= 1) {
    if (tid < off) sh[tid] = fmaxf(sh[tid], sh[tid + off]);
    __syncthreads();
  }
  mx = sh[0]; __syncthreads();
  float s = 0.0f;
#pragma unroll
  for (int j = 0; j < 4; ++j) { x[j] = __expf(x[j] - mx); s += x[j]; }
  sh[tid] = s; __syncthreads();
  for (int off = 128; off > 0; off >>= 1) {
    if (tid < off) sh[tid] += sh[tid + off];
    __syncthreads();
  }
  const float inv = 1.0f / sh[0];
#pragma unroll
  for (int j = 0; j < 4; ++j) {
    float o = x[j] * inv;
    p[tid + j * 256] = o;
    attn16[row * 1024 + tid + j * 256] = (_Float16)o;
  }
}

__global__ __launch_bounds__(256) void f2h_k(const float* __restrict__ in,
                                             _Float16* __restrict__ out, long n) {
  long i = (long)blockIdx.x * blockDim.x + threadIdx.x;
  if (i < n) out[i] = (_Float16)in[i];
}

// ---------------------------------------------------------------------------
extern "C" void kernel_launch(void* const* d_in, const int* in_sizes, int n_in,
                              void* d_out, int out_size, void* d_ws, size_t ws_size,
                              hipStream_t stream) {
  (void)in_sizes; (void)n_in; (void)out_size; (void)ws_size;
  const float* enc_input = (const float*)d_in[0];
  const int*   mask      = (const int*)  d_in[1];
  const float* ln1_g = (const float*)d_in[2];
  const float* ln1_b = (const float*)d_in[3];
  const float* wq = (const float*)d_in[4];
  const float* wk = (const float*)d_in[5];
  const float* wv = (const float*)d_in[6];
  const float* wo = (const float*)d_in[7];
  const float* ln2_g = (const float*)d_in[8];
  const float* ln2_b = (const float*)d_in[9];
  const float* w1 = (const float*)d_in[10];
  const float* b1 = (const float*)d_in[11];
  const float* w2 = (const float*)d_in[12];
  const float* b2 = (const float*)d_in[13];

  float* enc_out  = (float*)d_out;                    // [B, 1024, 1024]
  float* attn_out = enc_out + (long)BATCH * PB;       // [B, H, 1024, 1024]

  // Workspace carve-out
  char* wsp = (char*)d_ws; size_t off = 0;
  auto alloc = [&](size_t bytes) -> void* {
    void* p = wsp + off; off += (bytes + 255) & ~(size_t)255; return p;
  };
  _Float16* h16    = (_Float16*)alloc((size_t)BATCH * PB * 2);  // LN out (reused for LN2)
  _Float16* q16    = (_Float16*)alloc((size_t)BATCH * PB * 2);  // Q (reused for FFN mid)
  _Float16* k16    = (_Float16*)alloc((size_t)BATCH * PB * 2);  // K (normalized in place)
  _Float16* v16    = (_Float16*)alloc((size_t)BATCH * PB * 2);  // V
  _Float16* attn16 = (_Float16*)alloc((size_t)BHMM * 2);        // softmax probs f16
  _Float16* out16  = (_Float16*)alloc((size_t)BATCH * PB * 2);  // heads concat
  float*    x2     = (float*)   alloc((size_t)BATCH * PB * 4);  // post-attn residual
  _Float16* wq16 = (_Float16*)alloc(PB * 2);
  _Float16* wk16 = (_Float16*)alloc(PB * 2);
  _Float16* wv16 = (_Float16*)alloc(PB * 2);
  _Float16* wo16 = (_Float16*)alloc(PB * 2);
  _Float16* w116 = (_Float16*)alloc(PB * 2);
  _Float16* w216 = (_Float16*)alloc(PB * 2);
  double*   stats = (double*)alloc(64 * sizeof(double));

  dim3 blk(256);
  const long cvt_blocks = (PB + 255) / 256;
  f2h_k<<<cvt_blocks, blk, 0, stream>>>(wq, wq16, PB);
  f2h_k<<<cvt_blocks, blk, 0, stream>>>(wk, wk16, PB);
  f2h_k<<<cvt_blocks, blk, 0, stream>>>(wv, wv16, PB);
  f2h_k<<<cvt_blocks, blk, 0, stream>>>(wo, wo16, PB);
  f2h_k<<<cvt_blocks, blk, 0, stream>>>(w1, w116, PB);
  f2h_k<<<cvt_blocks, blk, 0, stream>>>(w2, w216, PB);

  // LN1
  ln_stats_k<<<BATCH, blk, 0, stream>>>(enc_input, stats, PB);
  ln_apply_k<<<4096, blk, 0, stream>>>(enc_input, stats, ln1_g, ln1_b, h16,
                                       PB, (long)BATCH * PB);

  // Q/K/V projections: [1024x1024] @ W^T per batch
  dim3 g_full(MDIM / 128, MDIM / 128, BATCH);
  GemmP p{};
  p.A = h16; p.lda = SDIM; p.aBatch = PB; p.aHead = 0;
  p.ldb = SDIM; p.bBatch = 0; p.bHead = 0;
  p.ldc = MDIM; p.cBatch = PB; p.cHead = 0;
  p.K = SDIM; p.zshift = 0;
  p.Bm = wq16; p.Ch = q16; gemm_wmma_k<true, 0><<<g_full, blk, 0, stream>>>(p);
  p.Bm = wk16; p.Ch = k16; gemm_wmma_k<true, 0><<<g_full, blk, 0, stream>>>(p);
  p.Bm = wv16; p.Ch = v16; gemm_wmma_k<true, 0><<<g_full, blk, 0, stream>>>(p);

  // Normalize K over head axis
  knorm_k<<<((long)BATCH * MDIM * DHEAD + 255) / 256, blk, 0, stream>>>(k16);

  // Attention scores per (b,h): Q [1024x128] @ Kn^T [128x1024], mask -> -1e9
  GemmP ps{};
  ps.A = q16; ps.lda = MDIM; ps.aBatch = PB; ps.aHead = DHEAD;
  ps.Bm = k16; ps.ldb = MDIM; ps.bBatch = PB; ps.bHead = DHEAD;
  ps.Cf = attn_out; ps.ldc = MDIM; ps.cBatch = (long)NHEAD * PB; ps.cHead = PB;
  ps.mask = mask; ps.maskBatch = PB;
  ps.K = DHEAD; ps.zshift = 3;
  dim3 g_sc(MDIM / 128, MDIM / 128, BATCH * NHEAD);
  gemm_wmma_k<true, 2><<<g_sc, blk, 0, stream>>>(ps);

  // Softmax rows: f32 final output + f16 copy for next GEMM
  softmax_row_k<<<(long)BATCH * NHEAD * MDIM, blk, 0, stream>>>(attn_out, attn16);

  // attn @ V per (b,h): [1024x1024] @ [1024x128] -> heads concatenated
  GemmP pav{};
  pav.A = attn16; pav.lda = MDIM; pav.aBatch = (long)NHEAD * PB; pav.aHead = PB;
  pav.Bm = v16; pav.ldb = MDIM; pav.bBatch = PB; pav.bHead = DHEAD;
  pav.Ch = out16; pav.ldc = MDIM; pav.cBatch = PB; pav.cHead = DHEAD;
  pav.K = MDIM; pav.zshift = 3;
  dim3 g_av(DHEAD / 128, MDIM / 128, BATCH * NHEAD);
  gemm_wmma_k<false, 0><<<g_av, blk, 0, stream>>>(pav);

  // Output projection + residual: x2 = out @ Wo^T + enc_input
  GemmP pwo{};
  pwo.A = out16; pwo.lda = MDIM; pwo.aBatch = PB; pwo.aHead = 0;
  pwo.Bm = wo16; pwo.ldb = MDIM; pwo.bBatch = 0; pwo.bHead = 0;
  pwo.Cf = x2; pwo.ldc = MDIM; pwo.cBatch = PB; pwo.cHead = 0;
  pwo.add = enc_input; pwo.addBatch = PB;
  pwo.K = MDIM; pwo.zshift = 0;
  gemm_wmma_k<true, 3><<<g_full, blk, 0, stream>>>(pwo);

  // LN2 (reuse h16)
  ln_stats_k<<<BATCH, blk, 0, stream>>>(x2, stats + 16, PB);
  ln_apply_k<<<4096, blk, 0, stream>>>(x2, stats + 16, ln2_g, ln2_b, h16,
                                       PB, (long)BATCH * PB);

  // FFN 1: mid = relu(h2 @ w1^T + b1)  (mid reuses q16)
  GemmP p1{};
  p1.A = h16; p1.lda = SDIM; p1.aBatch = PB; p1.aHead = 0;
  p1.Bm = w116; p1.ldb = SDIM; p1.bBatch = 0; p1.bHead = 0;
  p1.Ch = q16; p1.ldc = MDIM; p1.cBatch = PB; p1.cHead = 0;
  p1.bias = b1;
  p1.K = SDIM; p1.zshift = 0;
  gemm_wmma_k<true, 4><<<g_full, blk, 0, stream>>>(p1);

  // FFN 2 + bias + residual -> enc_output
  GemmP p2{};
  p2.A = q16; p2.lda = MDIM; p2.aBatch = PB; p2.aHead = 0;
  p2.Bm = w216; p2.ldb = MDIM; p2.bBatch = 0; p2.bHead = 0;
  p2.Cf = enc_out; p2.ldc = MDIM; p2.cBatch = PB; p2.cHead = 0;
  p2.bias = b2; p2.add = x2; p2.addBatch = PB;
  p2.K = MDIM; p2.zshift = 0;
  gemm_wmma_k<true, 5><<<g_full, blk, 0, stream>>>(p2);
}
